// MethodDeepGATResNet_84945863180849
// MI455X (gfx1250) — matrix operand order repaired and checked by reading, not compiled, not run on hardware
//
#include <hip/hip_runtime.h>
#include <math.h>

typedef __attribute__((ext_vector_type(16))) __bf16 v16bf;
typedef __attribute__((ext_vector_type(8)))  float  v8f;

#define ALPHA_SLOPE 0.2f
#define NEG_BIG     -9.0e15f

__device__ __forceinline__ __bf16 f2bf(float f) {
  unsigned u = __builtin_bit_cast(unsigned, f);
  u += 0x7FFFu + ((u >> 16) & 1u);            // round-to-nearest-even
  unsigned short h = (unsigned short)(u >> 16);
  return __builtin_bit_cast(__bf16, h);
}
__device__ __forceinline__ float bf2f(__bf16 b) {
  unsigned u = ((unsigned)__builtin_bit_cast(unsigned short, b)) << 16;
  return __builtin_bit_cast(float, u);
}

// A-fragment K base for 16-bit 16x32 A layout: VGPR v<4: K=2v+8g ; v>=4: K=16+2(v-4)+8g
__device__ __forceinline__ int a_kbase(int v, int g) {
  return (v < 4) ? (2 * v + 8 * g) : (16 + 2 * (v - 4) + 8 * g);
}

// ---- CDNA5 async global->LDS copies (ASYNCcnt path, CDNA5 ISA) ----
__device__ __forceinline__ void async_copy_b128(const void* lds_dst, const void* gsrc) {
  unsigned l = (unsigned)reinterpret_cast<unsigned long long>(lds_dst);   // LDS byte addr
  unsigned long long ga = reinterpret_cast<unsigned long long>(gsrc);
  asm volatile("global_load_async_to_lds_b128 %0, %1, off"
               :: "v"(l), "v"(ga) : "memory");
}
__device__ __forceinline__ void async_copy_b32(const void* lds_dst, const void* gsrc) {
  unsigned l = (unsigned)reinterpret_cast<unsigned long long>(lds_dst);
  unsigned long long ga = reinterpret_cast<unsigned long long>(gsrc);
  asm volatile("global_load_async_to_lds_b32 %0, %1, off"
               :: "v"(l), "v"(ga) : "memory");
}
__device__ __forceinline__ void wait_async() {
  asm volatile("s_wait_asynccnt 0x0" ::: "memory");
}

// ---------------------------------------------------------------------------
// bf16 WMMA GEMM: C tile = A[M,K](f32) x B[K,N](f32). One wave per block.
// BF16T=true: store bf16 TRANSPOSED [N-dim major]: C[(n)*ldc + m]  (ldc = M stride)
// BF16T=false: store f32 row-major: C[m*ldc + n]
// B columns >= bCols read as zero (pads OUT_F=40 -> 48).
// ---------------------------------------------------------------------------
template <bool BF16T>
__global__ __launch_bounds__(32)
void gemm_bf16_wmma(const float* __restrict__ A, int lda, long aHS,
                    const float* __restrict__ B, int ldb, int bCols, long bHS,
                    void* __restrict__ Cv, int ldc, long cHS, int K) {
  __shared__ alignas(16) __bf16 ldsA[16 * 32];    // [m][k]
  __shared__ alignas(16) __bf16 ldsBt[16 * 32];   // [n][k]
  const int lane = threadIdx.x;
  const int m = lane & 15, g = lane >> 4;
  const int m0 = blockIdx.x * 16;
  const int n0 = blockIdx.y * 16;
  const int head = blockIdx.z;
  A += (long)head * aHS;
  B += (long)head * bHS;

  v8f acc = {};
  const unsigned* ldsA32  = (const unsigned*)ldsA;
  const unsigned* ldsBt32 = (const unsigned*)ldsBt;

  for (int k0 = 0; k0 < K; k0 += 32) {
    {  // stage A tile 16x32, lane covers half a row
      const float* src = A + (long)(m0 + m) * lda + k0 + g * 16;
#pragma unroll
      for (int q = 0; q < 4; ++q) {
        float4 f = *(const float4*)(src + q * 4);
        int base = m * 32 + g * 16 + q * 4;
        ldsA[base + 0] = f2bf(f.x);
        ldsA[base + 1] = f2bf(f.y);
        ldsA[base + 2] = f2bf(f.z);
        ldsA[base + 3] = f2bf(f.w);
      }
    }
    {  // stage B tile 32x16 transposed, lane = k-row
      const float* src = B + (long)(k0 + lane) * ldb;
#pragma unroll
      for (int c = 0; c < 16; ++c) {
        int col = n0 + c;
        float v = (col < bCols) ? src[col] : 0.0f;
        ldsBt[c * 32 + lane] = f2bf(v);
      }
    }
    __syncthreads();
    union { v16bf v; unsigned u[8]; } af, bfr;
#pragma unroll
    for (int p = 0; p < 8; ++p) {
      af.u[p]  = ldsA32[m * 16 + (a_kbase(p, g) >> 1)];
      bfr.u[p] = ldsBt32[m * 16 + (a_kbase(p, g) >> 1)];
    }
    acc = __builtin_amdgcn_wmma_f32_16x16x32_bf16(false, af.v, false, bfr.v,
                                                  (short)0, acc, false, false);
    __syncthreads();
  }
  if constexpr (BF16T) {
    __bf16* C = (__bf16*)Cv + (long)head * cHS;
#pragma unroll
    for (int r = 0; r < 8; ++r)
      C[(long)(n0 + m) * ldc + m0 + r + 8 * g] = f2bf(acc[r]);
  } else {
    float* C = (float*)Cv + (long)head * cHS;
#pragma unroll
    for (int r = 0; r < 8; ++r)
      C[(long)(m0 + r + 8 * g) * ldc + n0 + m] = acc[r];
  }
}

// ---------------------------------------------------------------------------
// Flash-style GAT attention. 256 threads = 8 waves; block covers 128 rows,
// each wave a 16-row tile; all waves share one async-staged H/r j-tile.
// hT: bf16 TRANSPOSED features [Fp][N] (ldT = N). NT feature tiles of 16.
// ---------------------------------------------------------------------------
template <int NT>
__global__ __launch_bounds__(256)
void gat_attn_wmma(const __bf16* __restrict__ hTbase, long hHeadStride, int ldT,
                   const float* __restrict__ lscBase, const float* __restrict__ rscBase,
                   long lrHeadStride,
                   const int* __restrict__ adj,
                   const float* __restrict__ res, int resLd,
                   float* __restrict__ out, int outLd,
                   int headColStride, int applyElu, int N) {
  constexpr int Fp = NT * 16;
  __shared__ alignas(16) __bf16 ldsHt[Fp * 32];   // [col][k]
  __shared__ alignas(16) float  ldsR[32];
  const int tid = threadIdx.x;
  const int wave = tid >> 5, lane = tid & 31;
  const int m = lane & 15, g = lane >> 4;
  const int i0 = blockIdx.x * 128 + wave * 16;
  const int head = blockIdx.y;
  const __bf16* hT = hTbase + (long)head * hHeadStride;
  const float* lsc = lscBase + (long)head * lrHeadStride;
  const float* rsc = rscBase + (long)head * lrHeadStride;
  const int colOff = head * headColStride;

  const float l_m = lsc[i0 + m];
  float rowMax = -3.0e38f, rowSum = 0.0f;
  v8f acc[NT] = {};
  const unsigned* ldsHt32 = (const unsigned*)ldsHt;
  const long adjRow = (long)(i0 + m) * N;

  for (int j0 = 0; j0 < N; j0 += 32) {
    // ---- cooperative async staging of H^T tile (Fp x 32 bf16) + r scores ----
    if (tid < Fp * 4) {                // 16B segments: 4 per 32-wide bf16 row
      int col = tid >> 2, seg = tid & 3;
      async_copy_b128(&ldsHt[col * 32 + seg * 8], hT + (long)col * ldT + j0 + seg * 8);
    }
    if (tid < 32) async_copy_b32(&ldsR[tid], rsc + j0 + tid);
    wait_async();
    __syncthreads();

    __builtin_prefetch(adj + adjRow + j0 + 32, 0, 3);   // next adj tile

    // ---- scores for this lane's 16 (row m, K) slots in A-fragment order ----
    float e[16];
    float tmax = -3.0e38f;
#pragma unroll
    for (int v = 0; v < 8; ++v) {
      int kb = a_kbase(v, g);
      int2 ad = *(const int2*)(adj + adjRow + j0 + kb);
      float2 rr = *(const float2*)(&ldsR[kb]);
      float e0 = l_m + rr.x; e0 = e0 > 0.f ? e0 : ALPHA_SLOPE * e0;
      float e1 = l_m + rr.y; e1 = e1 > 0.f ? e1 : ALPHA_SLOPE * e1;
      e0 = (ad.x > 0) ? e0 : NEG_BIG;
      e1 = (ad.y > 0) ? e1 : NEG_BIG;
      e[2 * v] = e0; e[2 * v + 1] = e1;
      tmax = fmaxf(tmax, fmaxf(e0, e1));
    }
    tmax = fmaxf(tmax, __shfl_xor(tmax, 16, 32));
    float newMax = fmaxf(rowMax, tmax);
    float scale = __expf(rowMax - newMax);
    float p[16];
    float tsum = 0.f;
#pragma unroll
    for (int i = 0; i < 16; ++i) { p[i] = __expf(e[i] - newMax); tsum += p[i]; }
    tsum += __shfl_xor(tsum, 16, 32);
    rowSum = rowSum * scale + tsum;
    rowMax = newMax;

    // rescale accumulators: C-layout row r+8g gets scale from lane (r+8g)
#pragma unroll
    for (int r = 0; r < 8; ++r) {
      float sc = __shfl(scale, r + 8 * g, 32);
#pragma unroll
      for (int t = 0; t < NT; ++t) acc[t][r] *= sc;
    }
    // pack P into the bf16 A fragment
    union { v16bf v; unsigned u[8]; } pf;
#pragma unroll
    for (int v = 0; v < 8; ++v) {
      unsigned lo = (unsigned)__builtin_bit_cast(unsigned short, f2bf(p[2 * v]));
      unsigned hi = (unsigned)__builtin_bit_cast(unsigned short, f2bf(p[2 * v + 1]));
      pf.u[v] = lo | (hi << 16);
    }
#pragma unroll
    for (int t = 0; t < NT; ++t) {
      union { v16bf v; unsigned u[8]; } hf;
#pragma unroll
      for (int pp = 0; pp < 8; ++pp)
        hf.u[pp] = ldsHt32[(t * 16 + m) * 16 + (a_kbase(pp, g) >> 1)];
      acc[t] = __builtin_amdgcn_wmma_f32_16x16x32_bf16(false, pf.v, false, hf.v,
                                                       (short)0, acc[t], false, false);
    }
    __syncthreads();   // protect ldsHt/ldsR before next stage
  }
  // ---- epilogue: divide by row sum, ELU, residual, store ----
#pragma unroll
  for (int r = 0; r < 8; ++r) {
    int row = i0 + r + 8 * g;
    float inv = 1.0f / __shfl(rowSum, r + 8 * g, 32);
#pragma unroll
    for (int t = 0; t < NT; ++t) {
      float v = acc[t][r] * inv;
      if (applyElu) v = v > 0.f ? v : (__expf(v) - 1.0f);
      int col = colOff + t * 16 + m;
      if (res) v += res[(long)row * resLd + col];
      out[(long)row * outLd + col] = v;
    }
  }
}

// l = h . a[0:Fv], r = h . a[Fv:2Fv] per row, from transposed bf16 features
__global__ void attn_scores(const __bf16* __restrict__ hTbase, long hHeadStride, int ldT,
                            int Fv,
                            const float* __restrict__ aBase, long aHeadStride,
                            float* __restrict__ lsc, float* __restrict__ rsc,
                            long lrHeadStride, int N) {
  int row = blockIdx.x * blockDim.x + threadIdx.x;
  int head = blockIdx.y;
  if (row >= N) return;
  const __bf16* hT = hTbase + (long)head * hHeadStride;
  const float* a = aBase + (long)head * aHeadStride;
  float l = 0.f, r = 0.f;
  for (int f = 0; f < Fv; ++f) {
    float hv = bf2f(hT[(long)f * ldT + row]);
    l += hv * a[f];
    r += hv * a[Fv + f];
  }
  lsc[(long)head * lrHeadStride + row] = l;
  rsc[(long)head * lrHeadStride + row] = r;
}

// out = log_softmax(elu(hp) + resL) over 40 cols (inputs padded to 48)
__global__ void final_logsoftmax(const float* __restrict__ hp,
                                 const float* __restrict__ resL,
                                 float* __restrict__ outp, int N) {
  int row = blockIdx.x * blockDim.x + threadIdx.x;
  if (row >= N) return;
  const long b = (long)row * 48;
  float mx = -3.0e38f;
  for (int c = 0; c < 40; ++c) {
    float v = hp[b + c];
    v = v > 0.f ? v : (__expf(v) - 1.0f);
    v += resL[b + c];
    mx = fmaxf(mx, v);
  }
  float s = 0.f;
  for (int c = 0; c < 40; ++c) {
    float v = hp[b + c];
    v = v > 0.f ? v : (__expf(v) - 1.0f);
    v += resL[b + c];
    s += __expf(v - mx);
  }
  float lg = logf(s) + mx;
  for (int c = 0; c < 40; ++c) {
    float v = hp[b + c];
    v = v > 0.f ? v : (__expf(v) - 1.0f);
    v += resL[b + c];
    outp[(long)row * 40 + c] = v - lg;
  }
}

extern "C" void kernel_launch(void* const* d_in, const int* in_sizes, int n_in,
                              void* d_out, int out_size, void* d_ws, size_t ws_size,
                              hipStream_t stream) {
  (void)in_sizes; (void)n_in; (void)out_size; (void)ws_size;
  const float* x      = (const float*)d_in[0];
  const int*   adj    = (const int*)d_in[1];
  const float* Ws0    = (const float*)d_in[2];
  const float* as0    = (const float*)d_in[3];
  const float* Ws1    = (const float*)d_in[4];
  const float* as1    = (const float*)d_in[5];
  const float* W_out  = (const float*)d_in[6];
  const float* a_out  = (const float*)d_in[7];
  const float* R0     = (const float*)d_in[8];
  const float* R_last = (const float*)d_in[9];
  float* out = (float*)d_out;

  constexpr int N = 4096, IN_F = 512, HID = 64, NH = 4, Hc = 256;
  constexpr int OUT_F = 40, OUT_P = 48;

  float* ws   = (float*)d_ws;
  float* res0 = ws;  ws += (size_t)N * Hc;          // [4096,256] f32
  float* h0   = ws;  ws += (size_t)N * Hc;          // layer features (L0 out -> L1 in)
  float* hpA  = ws;  ws += (size_t)N * OUT_P;       // output-head attention result
  float* resL = ws;  ws += (size_t)N * OUT_P;       // res0 @ R_last (padded)
  float* lsc  = ws;  ws += (size_t)NH * N;
  float* rsc  = ws;  ws += (size_t)NH * N;
  __bf16* hprjT = (__bf16*)ws;                      // [NH][HID][N] bf16 transposed
  __bf16* hpoT  = hprjT + (size_t)NH * HID * N;     // [48][N] bf16 transposed

  dim3 wave(32), blk(256);

  // res0 = x @ R0   (f32 out)
  gemm_bf16_wmma<false><<<dim3(N / 16, Hc / 16, 1), wave, 0, stream>>>(
      x, IN_F, 0, R0, Hc, Hc, 0, res0, Hc, 0, IN_F);

  // ---- layer 0 ----
  gemm_bf16_wmma<true><<<dim3(N / 16, HID / 16, NH), wave, 0, stream>>>(
      x, IN_F, 0, Ws0, HID, HID, (long)IN_F * HID, hprjT, N, (long)HID * N, IN_F);
  attn_scores<<<dim3(N / 256, NH), blk, 0, stream>>>(
      hprjT, (long)HID * N, N, HID, as0, 2 * HID, lsc, rsc, N, N);
  gat_attn_wmma<4><<<dim3(N / 128, NH), blk, 0, stream>>>(
      hprjT, (long)HID * N, N, lsc, rsc, (long)N, adj, res0, Hc, h0, Hc, HID, 1, N);

  // ---- layer 1 (reads h0 -> hprjT, then overwrites h0) ----
  gemm_bf16_wmma<true><<<dim3(N / 16, HID / 16, NH), wave, 0, stream>>>(
      h0, Hc, 0, Ws1, HID, HID, (long)Hc * HID, hprjT, N, (long)HID * N, Hc);
  attn_scores<<<dim3(N / 256, NH), blk, 0, stream>>>(
      hprjT, (long)HID * N, N, HID, as1, 2 * HID, lsc, rsc, N, N);
  gat_attn_wmma<4><<<dim3(N / 128, NH), blk, 0, stream>>>(
      hprjT, (long)HID * N, N, lsc, rsc, (long)N, adj, res0, Hc, h0, Hc, HID, 1, N);

  // ---- output head (F=40 padded to 48) ----
  gemm_bf16_wmma<true><<<dim3(N / 16, OUT_P / 16, 1), wave, 0, stream>>>(
      h0, Hc, 0, W_out, OUT_F, OUT_F, 0, hpoT, N, 0, Hc);
  attn_scores<<<dim3(N / 256, 1), blk, 0, stream>>>(
      hpoT, 0, N, OUT_F, a_out, 0, lsc, rsc, 0, N);
  gat_attn_wmma<3><<<dim3(N / 128, 1), blk, 0, stream>>>(
      hpoT, 0, N, lsc, rsc, 0, adj, nullptr, 0, hpA, OUT_P, 0, 0, N);

  // resL = res0 @ R_last ; final elu + residual + log_softmax
  gemm_bf16_wmma<false><<<dim3(N / 16, OUT_P / 16, 1), wave, 0, stream>>>(
      res0, Hc, 0, R_last, OUT_F, OUT_F, 0, resL, OUT_P, 0, Hc);
  final_logsoftmax<<<dim3(N / 256), blk, 0, stream>>>(hpA, resL, out, N);
}
